// SPINN_56118042689580
// MI455X (gfx1250) — matrix operand order repaired and checked by reading, not compiled, not run on hardware
//
#include <hip/hip_runtime.h>
#include <hip/hip_bf16.h>
#include <math.h>

// ---------------- types ----------------
typedef __attribute__((ext_vector_type(16))) __bf16 v16bf;
typedef __attribute__((ext_vector_type(8)))  float  v8f;

// ---------------- problem constants ----------------
#define B_      512
#define S_      128
#define E_      300
#define H_      600
#define D_      300
#define T_      255
#define STK_    129          // S+1 stack slots
#define MROWS   65536        // B*S
#define EP_     320          // E padded to mult of 32
#define KT_W    10           // EP/32  k-tiles for word projection
#define NT_W_P  40           // n-tiles for word projection, padded to mult of 4 (608->640)
#define KCAT    640          // 2*EP (concatenated [lh|rh])
#define KT_C    20           // KCAT/32
#define NT_C_P  96           // n-tiles for gates, padded to mult of 4 (1504->1536)
#define LDG     1504         // gates row stride (padded 5*D)
#define FIVE_D  1500
#define BN_EPS  1e-5f

// ---------------- helpers ----------------
__device__ __forceinline__ unsigned f2bf_u(float x) {
    union { float f; unsigned u; } v; v.f = x;
    unsigned r = v.u + 0x7fffu + ((v.u >> 16) & 1u);   // round-to-nearest-even
    return r >> 16;
}
__device__ __forceinline__ unsigned packbf2(float lo, float hi) {
    return (f2bf_u(hi) << 16) | f2bf_u(lo);
}
// A-fragment K base per ISA table (16-bit A 16x32): VGPR j, half h
__device__ __forceinline__ int kbA(int j, int hf) {
    return ((j < 4) ? (2 * j) : (16 + 2 * (j - 4))) + 8 * hf;
}
// B-fragment K base (16-bit B 32x16): same K order as A (VGPR j, half h)
__device__ __forceinline__ int kbB(int j, int hf) { return ((j < 4) ? (2 * j) : (16 + 2 * (j - 4))) + 8 * hf; }

__device__ __forceinline__ float sigm(float x) { return 1.f / (1.f + expf(-x)); }

// ---------------- init ----------------
__global__ void zero_u32(unsigned* p, long n) {
    long i = (long)blockIdx.x * blockDim.x + threadIdx.x;
    if (i < n) p[i] = 0u;
}

// ---------------- weight / activation packing into WMMA fragment order ----------------
// sentence f32 [MROWS,300] -> A-frags [mt=4096][kt=10][lane=32][8 dwords]
__global__ void pack_sent(const float* __restrict__ sent, unsigned* __restrict__ frag) {
    int tid = blockIdx.x * blockDim.x + threadIdx.x;
    const int total = 4096 * KT_W * 32;
    if (tid >= total) return;
    int lane = tid & 31, kt = (tid >> 5) % KT_W, mt = tid / (32 * KT_W);
    int hf = lane >> 4, m = mt * 16 + (lane & 15);
    const float* row = sent + (size_t)m * E_;
    unsigned out[8];
#pragma unroll
    for (int j = 0; j < 8; ++j) {
        int k = kt * 32 + kbA(j, hf);
        float lo = (k     < E_) ? row[k]     : 0.f;
        float hi = (k + 1 < E_) ? row[k + 1] : 0.f;
        out[j] = packbf2(lo, hi);
    }
    unsigned* dst = frag + ((size_t)(mt * KT_W + kt) * 32 + lane) * 8;
#pragma unroll
    for (int j = 0; j < 8; ++j) dst[j] = out[j];
}

// W_word f32 [300,600] -> B-frags [nt=40][kt=10][lane][8]  (pad tiles are zero)
__global__ void pack_wword(const float* __restrict__ W, unsigned* __restrict__ frag) {
    int tid = blockIdx.x * blockDim.x + threadIdx.x;
    const int total = NT_W_P * KT_W * 32;
    if (tid >= total) return;
    int lane = tid & 31, kt = (tid >> 5) % KT_W, nt = tid / (32 * KT_W);
    int hf = lane >> 4, n = nt * 16 + (lane & 15);
    unsigned out[8];
#pragma unroll
    for (int j = 0; j < 8; ++j) {
        int k = kt * 32 + kbB(j, hf);
        float lo = (k     < E_ && n < H_) ? W[(size_t)k       * H_ + n] : 0.f;
        float hi = (k + 1 < E_ && n < H_) ? W[(size_t)(k + 1) * H_ + n] : 0.f;
        out[j] = packbf2(lo, hi);
    }
    unsigned* dst = frag + ((size_t)(nt * KT_W + kt) * 32 + lane) * 8;
#pragma unroll
    for (int j = 0; j < 8; ++j) dst[j] = out[j];
}

// [W_l;W_r] (each f32 [300,1500]) -> B-frags [nt=96][kt=20][lane][8], K rows 0..319=W_l, 320..639=W_r
__global__ void pack_wcat(const float* __restrict__ Wl, const float* __restrict__ Wr,
                          unsigned* __restrict__ frag) {
    int tid = blockIdx.x * blockDim.x + threadIdx.x;
    const int total = NT_C_P * KT_C * 32;
    if (tid >= total) return;
    int lane = tid & 31, kt = (tid >> 5) % KT_C, nt = tid / (32 * KT_C);
    int hf = lane >> 4, n = nt * 16 + (lane & 15);
    unsigned out[8];
#pragma unroll
    for (int j = 0; j < 8; ++j) {
        int k = kt * 32 + kbB(j, hf);
        const float* W = (k < EP_) ? Wl : Wr;
        int kk = (k < EP_) ? k : k - EP_;
        float lo = (kk     < D_ && n < FIVE_D) ? W[(size_t)kk       * FIVE_D + n] : 0.f;
        float hi = (kk + 1 < D_ && n < FIVE_D) ? W[(size_t)(kk + 1) * FIVE_D + n] : 0.f;
        out[j] = packbf2(lo, hi);
    }
    unsigned* dst = frag + ((size_t)(nt * KT_C + kt) * 32 + lane) * 8;
#pragma unroll
    for (int j = 0; j < 8; ++j) dst[j] = out[j];
}

// ---------------- WMMA GEMM over pre-packed fragments ----------------
// One wave computes a 16 x (16*NBLK) tile: the A fragment is loaded once per K-tile and
// reused against NBLK B fragments (NBLK wmma per NBLK+1 loads). 4 waves per block.
// NTP must be a multiple of NBLK; pad B-fragment tiles are zero so loads are safe,
// stores are guarded by n < nvalid.
template <int NBLK>
__global__ void gemm_frag(const unsigned* __restrict__ Af, const unsigned* __restrict__ Bf,
                          float* __restrict__ C, int KT, int NTP, int ldc, int nvalid,
                          const float* __restrict__ bias, int bias_n) {
    int wave = threadIdx.x >> 5, lane = threadIdx.x & 31;
    int nt0 = (blockIdx.x * 4 + wave) * NBLK;
    int mt = blockIdx.y;
    if (nt0 >= NTP) return;
    v8f acc[NBLK];
#pragma unroll
    for (int i = 0; i < NBLK; ++i) acc[i] = (v8f){0.f, 0.f, 0.f, 0.f, 0.f, 0.f, 0.f, 0.f};
    const unsigned* ap = Af + (size_t)mt * KT * 256 + lane * 8;
    const unsigned* bp = Bf + (size_t)nt0 * KT * 256 + lane * 8;
    for (int kt = 0; kt < KT; ++kt) {
        v16bf a = *(const v16bf*)ap;
#pragma unroll
        for (int i = 0; i < NBLK; ++i) {
            v16bf b = *(const v16bf*)(bp + (size_t)i * KT * 256);
            acc[i] = __builtin_amdgcn_wmma_f32_16x16x32_bf16(false, a, false, b,
                                                             (short)0, acc[i], false, false);
        }
        ap += 256; bp += 256;
    }
    int mbase = mt * 16 + (lane >> 4) * 8;       // C/D: VGPR i -> row i + 8*half
#pragma unroll
    for (int i = 0; i < NBLK; ++i) {
        int n = (nt0 + i) * 16 + (lane & 15);
        if (n < nvalid) {
            float bv = bias ? ((n < bias_n) ? bias[n] : 0.f) : 0.f;
#pragma unroll
            for (int r = 0; r < 8; ++r)
                C[(size_t)(mbase + r) * ldc + n] = acc[i][r] + bv;
        }
    }
}

// ---------------- BatchNorm ----------------
__global__ void bn_stats(const float* __restrict__ proj, float* __restrict__ sum,
                         float* __restrict__ sumsq) {
    int c = threadIdx.x;
    if (c >= H_) return;
    int r0 = blockIdx.x * 256;
    float s = 0.f, s2 = 0.f;
    for (int i = 0; i < 256; ++i) {
        float v = proj[(size_t)(r0 + i) * H_ + c];
        s += v; s2 += v * v;
    }
    atomicAdd(&sum[c], s);
    atomicAdd(&sumsq[c], s2);
}
__global__ void bn_finalize(const float* sum, const float* sumsq, const float* gamma,
                            const float* beta, float* scale, float* shiftv) {
    int c = threadIdx.x;
    if (c >= H_) return;
    const float invN = 1.f / (float)MROWS;
    float mean = sum[c] * invN;
    float var  = sumsq[c] * invN - mean * mean;
    float rstd = rsqrtf(var + BN_EPS);
    float sc = rstd * gamma[c];
    scale[c]  = sc;
    shiftv[c] = beta[c] - mean * sc;
}
__global__ void bn_apply(float* __restrict__ proj, const float* __restrict__ scale,
                         const float* __restrict__ shiftv) {
    long i = (long)blockIdx.x * blockDim.x + threadIdx.x;
    if (i >= (long)MROWS * H_) return;
    int c = (int)(i % H_);
    proj[i] = proj[i] * scale[c] + shiftv[c];
}

// ---------------- per-step scan kernels ----------------
// Build A = [lh | rh] (K=640, bf16 frags) from stack tops, per-batch pointers.
__global__ void gather_A(const float* __restrict__ stack_h, const int* __restrict__ sptr,
                         unsigned* __restrict__ frag) {
    int tid = blockIdx.x * blockDim.x + threadIdx.x;
    const int total = 32 * KT_C * 32;          // mt * kt * lane
    if (tid >= total) return;
    int lane = tid & 31, kt = (tid >> 5) % KT_C, mt = tid / (32 * KT_C);
    int hf = lane >> 4, b = mt * 16 + (lane & 15);
    int sp = sptr[b];
    int lpos = max(sp - 2, 0), rpos = max(sp - 1, 0);
    const float* lrow = stack_h + ((size_t)b * STK_ + lpos) * D_;
    const float* rrow = stack_h + ((size_t)b * STK_ + rpos) * D_;
    unsigned out[8];
#pragma unroll
    for (int j = 0; j < 8; ++j) {
        int k = kt * 32 + kbA(j, hf);
        float lo, hi;
        if (k < EP_) {
            lo = (k     < D_) ? lrow[k]     : 0.f;
            hi = (k + 1 < D_) ? lrow[k + 1] : 0.f;
        } else {
            int kk = k - EP_;
            lo = (kk     < D_) ? rrow[kk]     : 0.f;
            hi = (kk + 1 < D_) ? rrow[kk + 1] : 0.f;
        }
        out[j] = packbf2(lo, hi);
    }
    unsigned* dst = frag + ((size_t)(mt * KT_C + kt) * 32 + lane) * 8;
#pragma unroll
    for (int j = 0; j < 8; ++j) dst[j] = out[j];
}

// Tree-LSTM cell + masked stack writes (pointers NOT yet updated).
__global__ void cell_step(const float* __restrict__ gates, const float* __restrict__ proj,
                          const int* __restrict__ trans, const int* __restrict__ sptr,
                          const int* __restrict__ bptr, float* __restrict__ sh,
                          float* __restrict__ sc, int t) {
    int d = blockIdx.x * blockDim.x + threadIdx.x;
    if (d >= D_) return;
    int b = blockIdx.y;
    int tr = trans[(size_t)b * T_ + t];
    bool do_shift = (tr == 0), do_red = (tr == 1);
    int sp = sptr[b], bp = bptr[b];
    int lpos = max(sp - 2, 0), rpos = max(sp - 1, 0);
    size_t sb = (size_t)b * STK_;
    float lc = sc[(sb + lpos) * D_ + d];
    float rc = sc[(sb + rpos) * D_ + d];
    const float* g = gates + (size_t)b * LDG;
    float gi = g[d], gfl = g[D_ + d], gfr = g[2 * D_ + d], go = g[3 * D_ + d], gg = g[4 * D_ + d];
    float nc = sigm(gfl) * lc + sigm(gfr) * rc + sigm(gi) * tanhf(gg);
    float nh = sigm(go) * tanhf(nc);
    int widx = min(bp, S_ - 1);
    size_t wrow = ((size_t)b * S_ + widx) * H_;
    float wh = proj[wrow + d], wc = proj[wrow + D_ + d];
    int pos = do_shift ? sp : lpos;
    if (do_shift || do_red) {
        sh[(sb + pos) * D_ + d] = do_shift ? wh : nh;
        sc[(sb + pos) * D_ + d] = do_shift ? wc : nc;
    }
}

__global__ void ptr_update(const int* __restrict__ trans, int* __restrict__ sptr,
                           int* __restrict__ bptr, int t) {
    int b = blockIdx.x * blockDim.x + threadIdx.x;
    if (b >= B_) return;
    int tr = trans[(size_t)b * T_ + t];
    if (tr == 0)      { sptr[b] += 1; bptr[b] += 1; }
    else if (tr == 1) { sptr[b] -= 1; }
}

__global__ void finalize_out(const float* __restrict__ sh, const int* __restrict__ sptr,
                             float* __restrict__ out) {
    int d = blockIdx.x * blockDim.x + threadIdx.x;
    if (d >= D_) return;
    int b = blockIdx.y;
    int sp = max(sptr[b] - 1, 0);
    out[(size_t)b * D_ + d] = sh[((size_t)b * STK_ + sp) * D_ + d];
}

// ---------------- host orchestration ----------------
extern "C" void kernel_launch(void* const* d_in, const int* in_sizes, int n_in,
                              void* d_out, int out_size, void* d_ws, size_t ws_size,
                              hipStream_t stream) {
    const float* sentence    = (const float*)d_in[0];
    const int*   transitions = (const int*)  d_in[1];
    const float* W_word      = (const float*)d_in[2];
    const float* b_word      = (const float*)d_in[3];
    const float* gamma       = (const float*)d_in[4];
    const float* beta        = (const float*)d_in[5];
    const float* W_l         = (const float*)d_in[6];
    const float* W_r         = (const float*)d_in[7];
    const float* b_reduce    = (const float*)d_in[8];
    float* out = (float*)d_out;

    char* cur = (char*)d_ws;
    auto alloc = [&](size_t bytes) -> char* {
        char* p = cur;
        cur += (bytes + 255) & ~(size_t)255;
        return p;
    };
    // zero-initialized region first (stacks + stats + pointers)
    char* zbase = cur;
    float* stack_h = (float*)alloc((size_t)B_ * STK_ * D_ * 4);
    float* stack_c = (float*)alloc((size_t)B_ * STK_ * D_ * 4);
    float* bn_sum  = (float*)alloc(608 * 4);
    float* bn_sq   = (float*)alloc(608 * 4);
    int*   sptr    = (int*)  alloc(B_ * 4);
    int*   bptr    = (int*)  alloc(B_ * 4);
    long zero_dw = (long)((cur - zbase) / 4);
    // remaining scratch
    float*    bn_scale = (float*)alloc(608 * 4);
    float*    bn_shift = (float*)alloc(608 * 4);
    float*    proj     = (float*)alloc((size_t)MROWS * H_ * 4);
    unsigned* sentfrag = (unsigned*)alloc((size_t)4096   * KT_W * 256 * 4);
    unsigned* wwfrag   = (unsigned*)alloc((size_t)NT_W_P * KT_W * 256 * 4);
    unsigned* wcfrag   = (unsigned*)alloc((size_t)NT_C_P * KT_C * 256 * 4);
    unsigned* afrag    = (unsigned*)alloc((size_t)32     * KT_C * 256 * 4);
    float*    gates    = (float*)alloc((size_t)B_ * LDG * 4);
    (void)ws_size; (void)in_sizes; (void)n_in; (void)out_size;

    // 1) zero stacks / stats / pointers
    zero_u32<<<dim3((unsigned)((zero_dw + 255) / 256)), dim3(256), 0, stream>>>(
        (unsigned*)zbase, zero_dw);

    // 2) pack activations & weights into WMMA fragment order (bf16)
    pack_sent <<<dim3((4096 * KT_W * 32 + 255) / 256), dim3(256), 0, stream>>>(sentence, sentfrag);
    pack_wword<<<dim3((NT_W_P * KT_W * 32 + 255) / 256), dim3(256), 0, stream>>>(W_word, wwfrag);
    pack_wcat <<<dim3((NT_C_P * KT_C * 32 + 255) / 256), dim3(256), 0, stream>>>(W_l, W_r, wcfrag);

    // 3) word projection GEMM: proj = sentence @ W_word + b_word   [65536 x 600]
    //    NBLK=4 (16x64 per wave): M-parallelism is huge, so trade waves for A-reuse.
    {
        int wave_groups = NT_W_P / 4;                  // 10 waves of n-tiles per mt
        gemm_frag<4><<<dim3((wave_groups + 3) / 4, 4096), dim3(128), 0, stream>>>(
            sentfrag, wwfrag, proj, KT_W, NT_W_P, H_, H_, b_word, H_);
    }

    // 4) BatchNorm (training-mode batch stats), folded to scale/shift, applied in place
    bn_stats   <<<dim3(MROWS / 256), dim3(608), 0, stream>>>(proj, bn_sum, bn_sq);
    bn_finalize<<<dim3(1), dim3(608), 0, stream>>>(bn_sum, bn_sq, gamma, beta, bn_scale, bn_shift);
    bn_apply   <<<dim3(((unsigned)((long)MROWS * H_ + 255) / 256)), dim3(256), 0, stream>>>(
        proj, bn_scale, bn_shift);

    // 5) sequential shift-reduce scan: gather -> fused GEMM -> cell -> pointer update
    //    gates GEMM uses NBLK=2 (16x32 per wave): M=512 only, keep ~1.5k waves in flight.
    int gate_wave_groups = NT_C_P / 2;                 // 48 waves of n-tiles per mt
    dim3 gate_grid((gate_wave_groups + 3) / 4, 32);
    for (int t = 0; t < T_; ++t) {
        gather_A<<<dim3((32 * KT_C * 32 + 255) / 256), dim3(256), 0, stream>>>(
            stack_h, sptr, afrag);
        gemm_frag<2><<<gate_grid, dim3(128), 0, stream>>>(
            afrag, wcfrag, gates, KT_C, NT_C_P, LDG, LDG, b_reduce, FIVE_D);
        cell_step<<<dim3((D_ + 255) / 256, B_), dim3(256), 0, stream>>>(
            gates, proj, transitions, sptr, bptr, stack_h, stack_c, t);
        ptr_update<<<dim3((B_ + 255) / 256), dim3(256), 0, stream>>>(transitions, sptr, bptr, t);
    }

    // 6) read stack tops
    finalize_out<<<dim3((D_ + 255) / 256, B_), dim3(256), 0, stream>>>(stack_h, sptr, out);
}
